// ReservoirSTDPReadout_30631706755669
// MI455X (gfx1250) — matrix operand order repaired and checked by reading, not compiled, not run on hardware
//
#include <hip/hip_runtime.h>
#include <hip/hip_bf16.h>

typedef __attribute__((ext_vector_type(16))) _Float16 v16h;
typedef __attribute__((ext_vector_type(8)))  _Float16 h8;
typedef __attribute__((ext_vector_type(8)))  float    v8f;

#define B_    64
#define C_    128
#define T_    500
#define NRES  2048
#define NCLS  10
#define NWG   128      // persistent workgroups (each owns 16 neurons)
#define WGTH  128      // 4 waves of 32
#define NPW   16       // neurons per WG
#define LDS_STRIDE 2056  // 2048 + 8 halves pad -> 4-bank rotation per row

// ---------------- fragment loaders (contiguous k order, same for A and B) ----------
// A 16x32 f16: lane m=L%16, half h=L/16; contiguous 16 K values starting at 16h
__device__ __forceinline__ v16h load_fragA(const _Float16* __restrict__ p, int h) {
  union { v16h v; h8 x[2]; } u;
  u.x[0] = *(const h8*)(p + 16 * h);
  u.x[1] = *(const h8*)(p + 16 * h + 8);
  return u.v;
}
// B 32x16 f16: lane n=L%16, half h=L/16; contiguous 16 K values starting at 16h
__device__ __forceinline__ v16h load_fragB_lds(const _Float16* p, int h) {
  union { v16h v; h8 x[2]; } u;
  u.x[0] = *(const h8*)(p + 16 * h);
  u.x[1] = *(const h8*)(p + 16 * h + 8);
  return u.v;
}
__device__ __forceinline__ v16h load_fragB(const _Float16* __restrict__ p, int h) {
  return *(const v16h*)(p + 16 * h);
}

// ---------------- pre-pass kernels -----------------------------------------
__global__ __launch_bounds__(256) void k_cvt_f16(const float* __restrict__ in,
                                                 _Float16* __restrict__ out, size_t n) {
  size_t i = (size_t)blockIdx.x * 256 + threadIdx.x;
  if (i < n) out[i] = (_Float16)in[i];
}

// x (B,C,T) f32 -> xT (T,B,C) f16
__global__ __launch_bounds__(256) void k_xpose(const float* __restrict__ x,
                                               _Float16* __restrict__ xT) {
  size_t o = (size_t)blockIdx.x * 256 + threadIdx.x;   // exactly T*B*C threads
  int t = (int)(o / (B_ * C_));
  int r = (int)(o % (B_ * C_));
  int b = r >> 7;          // /128
  int c = r & (C_ - 1);
  xT[o] = (_Float16)x[(size_t)b * (C_ * T_) + (size_t)c * T_ + t];
}

__global__ __launch_bounds__(256) void k_zero_u32(unsigned* __restrict__ p, size_t n) {
  size_t i = (size_t)blockIdx.x * 256 + threadIdx.x;
  if (i < n) p[i] = 0u;
}

// ---------------- persistent SNN kernel ------------------------------------
__global__ __launch_bounds__(WGTH) void snn_persistent(
    const _Float16* __restrict__ xT,      // (T, 64, 128) f16
    const _Float16* __restrict__ Wi,      // (2048, 128)  f16
    const _Float16* __restrict__ Wr,      // (2048, 2048) f16
    const float*    __restrict__ Wc,      // (10, 2048)   f32
    _Float16* __restrict__ sbuf,          // 2 x (64 x 2048) f16, spike double buffer
    float* __restrict__ partials,         // 2 x NWG x 64 x 16 f32
    unsigned* __restrict__ bar,           // [0]=count [1]=generation
    float* __restrict__ out)              // 640 clf counts + 64*2048 res counts
{
  const int wg   = blockIdx.x;
  const int tid  = threadIdx.x;
  const int lane = tid & 31;
  const int w    = tid >> 5;              // wave id == M tile (0..3)
  const int hsel = lane >> 4;             // lane half
  const int nl   = lane & 15;             // local column within tile
  const int n0   = wg * NPW;              // this WG's neuron base

  __shared__ _Float16 wr_lds[NPW * LDS_STRIDE]; // 64.25 KB: this WG's W_res slice
  __shared__ float    s_lds[B_ * NPW];          // 4 KB: this step's spikes

  // ---- prologue: stage W_res slice into LDS once via gfx1250 async copy ----
  {
    unsigned lbase = (unsigned)(size_t)&wr_lds[0];
    #pragma unroll 4
    for (int i = 0; i < 32; ++i) {
      int c   = tid + WGTH * i;           // 0..4095 16-byte chunks
      int row = c >> 8;                   // 256 chunks per 2048-half row
      int col = c & 255;
      unsigned long long g =
          (unsigned long long)(size_t)(Wr + (size_t)(n0 + row) * NRES + col * 8);
      unsigned l = lbase + (unsigned)(row * (LDS_STRIDE * 2) + col * 16);
      asm volatile("global_load_async_to_lds_b128 %0, %1, off"
                   :: "v"(l), "v"(g) : "memory");
    }
    asm volatile("s_wait_asynccnt 0" ::: "memory");
  }
  __syncthreads();

  // persistent per-lane state: membrane + spike count for 8 (m,n) cells
  v8f vr, rc;
  #pragma unroll
  for (int i = 0; i < 8; ++i) { vr[i] = 0.f; rc[i] = 0.f; }
  // classifier state (only WG0's copy is used): 5 (b,class) slots per thread
  float vc[5], cc[5];
  #pragma unroll
  for (int j = 0; j < 5; ++j) { vc[j] = 0.f; cc[j] = 0.f; }

  const int mrow = 16 * w + nl;           // A row (batch) this lane fetches
  const _Float16* bRow  = &wr_lds[nl * LDS_STRIDE];   // B rows from LDS
  const _Float16* WiRow = Wi + (size_t)(n0 + nl) * C_;

  for (int t = 0; t < T_; ++t) {
    const _Float16* sprev = sbuf + (size_t)(t & 1) * (B_ * NRES);
    _Float16*       snew  = sbuf + (size_t)((t + 1) & 1) * (B_ * NRES);
    const _Float16* xrow  = xT + (size_t)t * (B_ * C_);

    v8f acc;
    #pragma unroll
    for (int i = 0; i < 8; ++i) acc[i] = 0.f;

    // Recurrent GEMM: I += s_prev @ W_res.T  (K = 2048); A global, B from LDS
    const _Float16* aBase = sprev + (size_t)mrow * NRES;
    #pragma unroll 4
    for (int k = 0; k < NRES; k += 32) {
      v16h a = load_fragA(aBase + k, hsel);
      v16h b = load_fragB_lds(bRow + k, hsel);
      acc = __builtin_amdgcn_wmma_f32_16x16x32_f16(false, a, false, b,
                                                   (short)0, acc, false, false);
    }
    // Feed-forward GEMM: I += x_t @ W_in.T  (K = 128); B read from global each step
    const _Float16* aBase2 = xrow + (size_t)mrow * C_;
    #pragma unroll
    for (int k = 0; k < C_; k += 32) {
      v16h a = load_fragA(aBase2 + k, hsel);
      v16h b = load_fragB(WiRow + k, hsel);
      acc = __builtin_amdgcn_wmma_f32_16x16x32_f16(false, a, false, b,
                                                   (short)0, acc, false, false);
    }

    // Prefetch next timestep's input tile (128 threads * 128B = full x_t)
    if (t + 1 < T_)
      __builtin_prefetch(xT + (size_t)(t + 1) * (B_ * C_) + tid * 64, 0, 0);

    // LIF update on D tile: lane holds (m = 16w + 8h + v, n = nl)
    #pragma unroll
    for (int v = 0; v < 8; ++v) {
      int m = 16 * w + 8 * hsel + v;
      float vv = 0.9f * vr[v] + acc[v];
      float sp = (vv >= 1.0f) ? 1.0f : 0.0f;
      rc[v] += sp;
      vr[v] = vv * (1.0f - sp);
      snew[(size_t)m * NRES + n0 + nl] = (_Float16)sp;
      s_lds[m * NPW + nl] = sp;
    }
    __syncthreads();

    // Classifier partial for this WG's 16 neurons: 64x10 dense dot (deterministic)
    float* pp = partials + (size_t)(((t & 1) * NWG + wg) * B_) * 16;
    #pragma unroll
    for (int j = 0; j < 5; ++j) {
      int oi = tid + WGTH * j;            // 0..639 exactly
      int bb = oi / 10, cl = oi % 10;
      const float* wrow = Wc + (size_t)cl * NRES + n0;
      float s = 0.f;
      #pragma unroll
      for (int jn = 0; jn < NPW; ++jn) s += s_lds[bb * NPW + jn] * wrow[jn];
      pp[bb * 16 + cl] = s;
    }

    // ---- grid-wide barrier (one per step) ----
    __threadfence();
    __syncthreads();
    if (tid == 0) {
      unsigned prev = atomicAdd(&bar[0], 1u);
      if (prev == NWG - 1) {
        atomicExch(&bar[0], 0u);
        __threadfence();
        atomicAdd(&bar[1], 1u);
      } else {
        while (atomicAdd(&bar[1], 0u) < (unsigned)(t + 1))
          __builtin_amdgcn_s_sleep(1);
      }
    }
    __syncthreads();
    __threadfence();

    // WG0 finalizes classifier LIF for step t (fixed-order sum over WGs)
    if (wg == 0) {
      const float* pr = partials + (size_t)((t & 1) * NWG) * B_ * 16;
      #pragma unroll
      for (int j = 0; j < 5; ++j) {
        int oi = tid + WGTH * j;
        int bb = oi / 10, cl = oi % 10;
        float ic = 0.f;
        for (int g = 0; g < NWG; ++g)
          ic += pr[(size_t)g * B_ * 16 + bb * 16 + cl];
        float v2 = 0.9f * vc[j] + ic;
        float c = (v2 >= 1.0f) ? 1.0f : 0.0f;
        cc[j] += c;
        vc[j] = v2 * (1.0f - c);
      }
    }
  }

  // Epilogue: reservoir spike counts (B x 2048), offset 640
  #pragma unroll
  for (int v = 0; v < 8; ++v) {
    int m = 16 * w + 8 * hsel + v;
    out[640 + (size_t)m * NRES + n0 + nl] = rc[v];
  }
  // Classifier spike counts (B x 10), first 640 elements
  if (wg == 0) {
    #pragma unroll
    for (int j = 0; j < 5; ++j) out[tid + WGTH * j] = cc[j];
  }
}

// ---------------- host launcher --------------------------------------------
extern "C" void kernel_launch(void* const* d_in, const int* in_sizes, int n_in,
                              void* d_out, int out_size, void* d_ws, size_t ws_size,
                              hipStream_t stream) {
  const float* x  = (const float*)d_in[0];   // (64,128,500)
  const float* Wi = (const float*)d_in[1];   // (2048,128)
  const float* Wr = (const float*)d_in[2];   // (2048,2048)
  const float* Wc = (const float*)d_in[3];   // (10,2048)
  float* out = (float*)d_out;

  char* ws = (char*)d_ws;
  size_t off = 0;
  auto alloc = [&](size_t bytes) -> char* {
    char* p = ws + off;
    off = (off + bytes + 255) & ~(size_t)255;
    return p;
  };
  _Float16* Wr16 = (_Float16*)alloc((size_t)NRES * NRES * 2);   // 8 MB
  _Float16* Wi16 = (_Float16*)alloc((size_t)NRES * C_ * 2);     // 0.5 MB
  _Float16* xT   = (_Float16*)alloc((size_t)T_ * B_ * C_ * 2);  // 8 MB
  _Float16* sbuf = (_Float16*)alloc((size_t)2 * B_ * NRES * 2); // 0.5 MB
  float* partials = (float*)alloc((size_t)2 * NWG * B_ * 16 * 4); // 1 MB
  unsigned* bar   = (unsigned*)alloc(256);

  // convert weights to f16
  k_cvt_f16<<<(NRES * NRES) / 256, 256, 0, stream>>>(Wr, Wr16, (size_t)NRES * NRES);
  k_cvt_f16<<<(NRES * C_) / 256, 256, 0, stream>>>(Wi, Wi16, (size_t)NRES * C_);
  // transpose x to (T,B,C) f16
  k_xpose<<<(T_ * B_ * C_) / 256, 256, 0, stream>>>(x, xT);
  // zero spike double-buffer and barrier state (fresh every call)
  k_zero_u32<<<((2 * B_ * NRES * 2 / 4) + 255) / 256, 256, 0, stream>>>(
      (unsigned*)sbuf, (size_t)(2 * B_ * NRES * 2 / 4));
  k_zero_u32<<<1, 256, 0, stream>>>(bar, (size_t)64);

  // persistent cooperative kernel: 128 WGs x 128 threads
  snn_persistent<<<NWG, WGTH, 0, stream>>>(xT, Wi16, Wr16, Wc,
                                           sbuf, partials, bar, out);
}
